// TGIB_53652731462296
// MI455X (gfx1250) — matrix operand order repaired and checked by reading, not compiled, not run on hardware
//
#include <hip/hip_runtime.h>
#include <hip/hip_bf16.h>

typedef __bf16 bf16;
typedef bf16 v16bf __attribute__((ext_vector_type(16)));
typedef float v8f  __attribute__((ext_vector_type(8)));

#define BB   2048
#define NN   64
#define FEAT 172
#define DD   516      // model dim
#define DK   258
#define KP_D 544      // 516 padded to mult of 32
#define KP_M 704      // 688 padded
#define KP_H 192      // 172 padded

// ---------------------------------------------------------------------------
// prep: effective score vectors  wq_eff[h][c] = sum_d Wq[h*258+d, c] * w1[d]
// ---------------------------------------------------------------------------
__global__ void prep_eff(const float* __restrict__ Wq, const float* __restrict__ Wk,
                         const float* __restrict__ wmap,
                         float* __restrict__ wqe, float* __restrict__ wke) {
    int i = blockIdx.x * blockDim.x + threadIdx.x;
    if (i >= 2 * 1032) return;
    int which = i / 1032;            // 0 = Q, 1 = K
    int r = i % 1032;
    int h = r / DD, c = r % DD;
    const float* Wm = which ? Wk : Wq;
    const float* wv = wmap + (which ? DK : 0);
    float s = 0.f;
    for (int d = 0; d < DK; ++d) s += Wm[(h * DK + d) * DD + c] * wv[d];
    (which ? wke : wqe)[h * DD + c] = s;
}

// convert f32 [R][K] -> bf16 [R][Kp], zero-padded
__global__ void cvt_pad(const float* __restrict__ src, bf16* __restrict__ dst,
                        int K, int Kp, int total) {
    int i = blockIdx.x * blockDim.x + threadIdx.x;
    if (i >= total) return;
    int r = i / Kp, k = i % Kp;
    dst[i] = (k < K) ? (bf16)src[r * K + k] : (bf16)0.f;
}

// zero pad columns [K,Kp) of bf16 [R][Kp]
__global__ void pad_zero(bf16* __restrict__ p, int K, int Kp, int R) {
    int w = Kp - K;
    int i = blockIdx.x * blockDim.x + threadIdx.x;
    if (i >= R * w) return;
    int r = i / w, j = i % w;
    p[r * Kp + K + j] = (bf16)0.f;
}

// zero-fill a bf16 buffer (dummy row for out-of-range B lanes)
__global__ void zero_fill(bf16* __restrict__ p, int n) {
    int i = blockIdx.x * blockDim.x + threadIdx.x;
    if (i < n) p[i] = (bf16)0.f;
}

// ---------------------------------------------------------------------------
// Attention: one block per batch row. Stages k_in[64][516] in LDS (bf16),
// computes scores via effective vectors, masked softmax, attention weights
// (written to output), and per-head context -> ctx_bf16 [B][2][544].
// ---------------------------------------------------------------------------
__global__ __launch_bounds__(256) void attn_kernel(
    const float* __restrict__ src, const float* __restrict__ src_t,
    const float* __restrict__ seq, const float* __restrict__ seq_t,
    const float* __restrict__ seq_e, const unsigned char* __restrict__ mask,
    const float* __restrict__ wq_eff, const float* __restrict__ wk_eff,
    float* __restrict__ qin_out,     // [B][516]
    bf16*  __restrict__ ctx_out,     // [B][2][544]
    float* __restrict__ attn_out)    // [2][B][64]
{
    __shared__ bf16  kin[NN * DD];   // 66 KB
    __shared__ float qin[DD];
    __shared__ float ssc[2][NN];
    __shared__ float attnW[2][NN];
    __shared__ float sred[4];

    const int b = blockIdx.x, tid = threadIdx.x;

    for (int i = tid; i < NN * FEAT; i += 256) {
        int n = i / FEAT, j = i % FEAT;
        int g = (b * NN + n) * FEAT + j;
        kin[n * DD + j]            = (bf16)seq[g];
        kin[n * DD + FEAT + j]     = (bf16)seq_e[g];
        kin[n * DD + 2 * FEAT + j] = (bf16)seq_t[g];
    }
    for (int c = tid; c < DD; c += 256) {
        float v = (c < FEAT) ? src[b * FEAT + c]
                 : (c < 2 * FEAT) ? 0.f
                 : src_t[b * FEAT + (c - 2 * FEAT)];
        qin[c] = v;
        qin_out[b * DD + c] = v;
    }
    __syncthreads();

    if (tid < 128) {                         // per-(n,h) key score
        int n = tid >> 1, h = tid & 1;
        const float* wk = wk_eff + h * DD;
        float s = 0.f;
        for (int c = 0; c < DD; ++c) s += (float)kin[n * DD + c] * wk[c];
        ssc[h][n] = s;
    } else if (tid < 130) {                  // per-head query score
        int h = tid - 128;
        const float* wq = wq_eff + h * DD;
        float s = 0.f;
        for (int c = 0; c < DD; ++c) s += qin[c] * wq[c];
        sred[h] = s;
    }
    __syncthreads();

    if (tid < 2) {                           // softmax stats per head
        int h = tid;
        float sq = sred[h], mx = -3.4e38f;
        for (int n = 0; n < NN; ++n) {
            float s = mask[b * NN + n] ? -1e10f : (sq + ssc[h][n]);
            ssc[h][n] = s;
            mx = fmaxf(mx, s);
        }
        float sum = 0.f;
        for (int n = 0; n < NN; ++n) sum += __expf(ssc[h][n] - mx);
        sred[h]     = mx;
        sred[2 + h] = 1.f / sum;
    }
    __syncthreads();

    if (tid < 128) {
        int n = tid >> 1, h = tid & 1;
        float a = __expf(ssc[h][n] - sred[h]) * sred[2 + h];
        attnW[h][n] = a;
        attn_out[(h * BB + b) * NN + n] = a;
    }
    __syncthreads();

    for (int p = tid; p < 2 * DD; p += 256) {   // ctx[h][c]
        int h = p / DD, c = p % DD;
        float acc = 0.f;
        for (int n = 0; n < NN; ++n) acc += attnW[h][n] * (float)kin[n * DD + c];
        ctx_out[(b * 2 + h) * KP_D + c] = (bf16)acc;
    }
    for (int p = tid; p < 2 * (KP_D - DD); p += 256) {
        int h = p / (KP_D - DD), j = p % (KP_D - DD);
        ctx_out[(b * 2 + h) * KP_D + DD + j] = (bf16)0.f;
    }
}

// ---------------------------------------------------------------------------
// Generic bf16 WMMA GEMM: C[m,n] = epi( sum_k A[m,k]*W[n,k] + bias[n] )
// 128 threads = 4 waves; each wave owns a 64x16 tile (4 M-frags share one
// B frag -> 4 WMMA per weight fragment). Block covers 256 rows x 16 cols.
// Out-of-range N lanes read a zero dummy row (no exec-masked loads).
// Exact gfx1250 16-bit fragment swizzles: A lane-half K interleave
// (chunks at k0+hi*8 and k0+16+hi*8), B lane = column, same K interleave as A.
// ---------------------------------------------------------------------------
union FragAB { v16bf v; uint4 q[2]; };

__global__ __launch_bounds__(128) void gemm_wmma(
    const bf16* __restrict__ A, int lda,
    const bf16* __restrict__ W, int ldw,
    const bf16* __restrict__ zrow,
    const float* __restrict__ bias,
    float* __restrict__ outF, int ldF,
    bf16*  __restrict__ outB, int ldB,
    int colOff, int Nact, int Kp, int epi)
{
    const int lane = threadIdx.x & 31;
    const int wave = threadIdx.x >> 5;
    const int hi = lane >> 4, lo = lane & 15;
    const int mBase = blockIdx.y * 256 + wave * 64;
    const int nBase = blockIdx.x * 16;

    const int  nCol = nBase + lo;
    const bool nOK  = nCol < Nact;
    const bf16* wRow = nOK ? (W + nCol * ldw) : zrow;   // unconditional loads

    const bf16* aRow0 = A + (mBase + lo) * lda;
    const int mStep = 16 * lda;

    v8f acc0 = {0.f,0.f,0.f,0.f,0.f,0.f,0.f,0.f};
    v8f acc1 = acc0, acc2 = acc0, acc3 = acc0;

    for (int k0 = 0; k0 < Kp; k0 += 32) {
        FragAB bfr;
        bfr.q[0] = *(const uint4*)(wRow + k0 + hi * 8);       // K = hi*8 + 0..7
        bfr.q[1] = *(const uint4*)(wRow + k0 + hi * 8 + 16);  // K = 16+hi*8+0..7

        FragAB a0, a1, a2, a3;
        const bf16* ap = aRow0 + k0 + hi * 8;
        a0.q[0] = *(const uint4*)(ap);                        // K = hi*8 + 0..7
        a0.q[1] = *(const uint4*)(ap + 16);                   // K = 16+hi*8+0..7
        a1.q[0] = *(const uint4*)(ap + mStep);
        a1.q[1] = *(const uint4*)(ap + mStep + 16);
        a2.q[0] = *(const uint4*)(ap + 2 * mStep);
        a2.q[1] = *(const uint4*)(ap + 2 * mStep + 16);
        a3.q[0] = *(const uint4*)(ap + 3 * mStep);
        a3.q[1] = *(const uint4*)(ap + 3 * mStep + 16);

        acc0 = __builtin_amdgcn_wmma_f32_16x16x32_bf16(false, a0.v, false, bfr.v, (short)0, acc0, false, false);
        acc1 = __builtin_amdgcn_wmma_f32_16x16x32_bf16(false, a1.v, false, bfr.v, (short)0, acc1, false, false);
        acc2 = __builtin_amdgcn_wmma_f32_16x16x32_bf16(false, a2.v, false, bfr.v, (short)0, acc2, false, false);
        acc3 = __builtin_amdgcn_wmma_f32_16x16x32_bf16(false, a3.v, false, bfr.v, (short)0, acc3, false, false);
    }

    if (!nOK) return;
    const float bv = bias ? bias[nCol] : 0.f;
    const int nc = colOff + nCol;
    v8f accs[4] = {acc0, acc1, acc2, acc3};
    #pragma unroll
    for (int t = 0; t < 4; ++t) {
        #pragma unroll
        for (int r = 0; r < 8; ++r) {
            float v = accs[t][r] + bv;
            if (epi == 1) v = (v > 0.f) ? v : 0.2f * v;   // leaky relu 0.2
            else if (epi == 2) v = fmaxf(v, 0.f);         // relu
            int m = mBase + t * 16 + hi * 8 + r;
            if (outF) outF[m * ldF + nc] = v;
            if (outB) outB[m * ldB + nc] = (bf16)v;
        }
    }
}

// ---------------------------------------------------------------------------
// LayerNorm(fc_out + q_in) -> xcat[b][0:516];  xcat[516:688]=src;  pad->704
// ---------------------------------------------------------------------------
__global__ __launch_bounds__(256) void ln_kernel(
    const float* __restrict__ fc_out, const float* __restrict__ qin,
    const float* __restrict__ src,
    const float* __restrict__ g, const float* __restrict__ be,
    bf16* __restrict__ xcat)
{
    __shared__ float xrow[DD];
    __shared__ float rs[256], rss[256];
    const int b = blockIdx.x, tid = threadIdx.x;

    float s = 0.f, ss = 0.f;
    for (int c = tid; c < DD; c += 256) {
        float x = fc_out[b * DD + c] + qin[b * DD + c];
        xrow[c] = x; s += x; ss += x * x;
    }
    rs[tid] = s; rss[tid] = ss;
    __syncthreads();
    for (int off = 128; off > 0; off >>= 1) {
        if (tid < off) { rs[tid] += rs[tid + off]; rss[tid] += rss[tid + off]; }
        __syncthreads();
    }
    const float mu  = rs[0] * (1.f / DD);
    const float var = rss[0] * (1.f / DD) - mu * mu;
    const float rstd = rsqrtf(var + 1e-5f);

    for (int c = tid; c < DD; c += 256)
        xcat[b * KP_M + c] = (bf16)((xrow[c] - mu) * rstd * g[c] + be[c]);
    for (int c = tid; c < FEAT; c += 256)
        xcat[b * KP_M + DD + c] = (bf16)src[b * FEAT + c];
    if (tid < KP_M - (DD + FEAT))
        xcat[b * KP_M + DD + FEAT + tid] = (bf16)0.f;
}

// ---------------------------------------------------------------------------
static inline size_t alignUp(size_t x) { return (x + 255) & ~size_t(255); }

extern "C" void kernel_launch(void* const* d_in, const int* in_sizes, int n_in,
                              void* d_out, int out_size, void* d_ws, size_t ws_size,
                              hipStream_t stream) {
    const float* src   = (const float*)d_in[0];
    const float* src_t = (const float*)d_in[1];
    const float* seq   = (const float*)d_in[2];
    const float* seq_t = (const float*)d_in[3];
    const float* seq_e = (const float*)d_in[4];
    const unsigned char* mask = (const unsigned char*)d_in[5];
    const float* Wq   = (const float*)d_in[6];
    const float* Wk   = (const float*)d_in[7];
    const float* Wv   = (const float*)d_in[8];
    const float* wmap = (const float*)d_in[9];
    const float* fc_W = (const float*)d_in[10];
    const float* fc_b = (const float*)d_in[11];
    const float* ln_g = (const float*)d_in[12];
    const float* ln_b = (const float*)d_in[13];
    const float* m1_W = (const float*)d_in[14];
    const float* m1_b = (const float*)d_in[15];
    const float* m2_W = (const float*)d_in[16];
    const float* m2_b = (const float*)d_in[17];

    float* out2     = (float*)d_out;                 // [2048][172]
    float* attn_out = (float*)d_out + BB * FEAT;     // [2][2048][64]

    // carve workspace
    char* p = (char*)d_ws;
    float* wqe   = (float*)p;  p += alignUp(1032 * 4);
    float* wke   = (float*)p;  p += alignUp(1032 * 4);
    float* qin   = (float*)p;  p += alignUp((size_t)BB * DD * 4);
    bf16*  ctx   = (bf16*)p;   p += alignUp((size_t)BB * 2 * KP_D * 2);
    bf16*  Wv_b  = (bf16*)p;   p += alignUp((size_t)DD * KP_D * 2);
    bf16*  fc_b16= (bf16*)p;   p += alignUp((size_t)DD * KP_D * 2);
    bf16*  m1_b16= (bf16*)p;   p += alignUp((size_t)FEAT * KP_M * 2);
    bf16*  m2_b16= (bf16*)p;   p += alignUp((size_t)FEAT * KP_H * 2);
    bf16*  outpre= (bf16*)p;   p += alignUp((size_t)BB * KP_D * 2);
    float* fcout = (float*)p;  p += alignUp((size_t)BB * DD * 4);
    bf16*  xcat  = (bf16*)p;   p += alignUp((size_t)BB * KP_M * 2);
    bf16*  h1    = (bf16*)p;   p += alignUp((size_t)BB * KP_H * 2);
    bf16*  zrow  = (bf16*)p;   p += alignUp((size_t)KP_M * 2);
    (void)ws_size; (void)in_sizes; (void)n_in; (void)out_size;

    // 1) effective score vectors + zero dummy row
    prep_eff<<<(2 * 1032 + 255) / 256, 256, 0, stream>>>(Wq, Wk, wmap, wqe, wke);
    zero_fill<<<(KP_M + 255) / 256, 256, 0, stream>>>(zrow, KP_M);

    // 2) weight conversions (f32 -> bf16, K zero-padded)
    {
        int t1 = DD * KP_D;
        cvt_pad<<<(t1 + 255) / 256, 256, 0, stream>>>(Wv,   Wv_b,   DD, KP_D, t1);
        cvt_pad<<<(t1 + 255) / 256, 256, 0, stream>>>(fc_W, fc_b16, DD, KP_D, t1);
        int t2 = FEAT * KP_M;
        cvt_pad<<<(t2 + 255) / 256, 256, 0, stream>>>(m1_W, m1_b16, DD + FEAT, KP_M, t2);
        int t3 = FEAT * KP_H;
        cvt_pad<<<(t3 + 255) / 256, 256, 0, stream>>>(m2_W, m2_b16, FEAT, KP_H, t3);
    }
    // activation pad columns (deterministic every call)
    pad_zero<<<(BB * (KP_D - DD) + 255) / 256, 256, 0, stream>>>(outpre, DD, KP_D, BB);
    pad_zero<<<(BB * (KP_H - FEAT) + 255) / 256, 256, 0, stream>>>(h1, FEAT, KP_H, BB);

    // 3) attention (single pass over seq tensors, LDS-staged)
    attn_kernel<<<BB, 256, 0, stream>>>(src, src_t, seq, seq_t, seq_e, mask,
                                        wqe, wke, qin, ctx, attn_out);

    // 4) out_pre[:, h*258 + d] = ctx[:,h,:] @ Wv[h*258+d,:]   (two WMMA GEMMs)
    for (int h = 0; h < 2; ++h) {
        gemm_wmma<<<dim3((DK + 15) / 16, BB / 256), 128, 0, stream>>>(
            ctx + h * KP_D, 2 * KP_D,
            Wv_b + h * DK * KP_D, KP_D, zrow,
            nullptr, nullptr, 0, outpre, KP_D,
            h * DK, DK, KP_D, 0);
    }

    // 5) fc: leaky_relu(out_pre @ fc_W.T + fc_b) -> fcout (f32)
    gemm_wmma<<<dim3((DD + 15) / 16, BB / 256), 128, 0, stream>>>(
        outpre, KP_D, fc_b16, KP_D, zrow, fc_b,
        fcout, DD, nullptr, 0, 0, DD, KP_D, 1);

    // 6) layernorm(fcout + qin) and build [ln || src] concat (bf16, padded)
    ln_kernel<<<BB, 256, 0, stream>>>(fcout, qin, src, ln_g, ln_b, xcat);

    // 7) merge fc1: relu(xcat @ m1_W.T + m1_b) -> h1 (bf16)
    gemm_wmma<<<dim3((FEAT + 15) / 16, BB / 256), 128, 0, stream>>>(
        xcat, KP_M, m1_b16, KP_M, zrow, m1_b,
        nullptr, 0, h1, KP_H, 0, FEAT, KP_M, 2);

    // 8) merge fc2: h1 @ m2_W.T + m2_b -> out2 (f32, final output)
    gemm_wmma<<<dim3((FEAT + 15) / 16, BB / 256), 128, 0, stream>>>(
        h1, KP_H, m2_b16, KP_H, zrow, m2_b,
        out2, FEAT, nullptr, 0, 0, FEAT, KP_H, 0);
}
